// QuantLinear_74045236183241
// MI455X (gfx1250) — matrix operand order repaired and checked by reading, not compiled, not run on hardware
//
#include <hip/hip_runtime.h>

typedef _Float16 v16h __attribute__((ext_vector_type(16)));
typedef _Float16 v8h  __attribute__((ext_vector_type(8)));
typedef _Float16 h2   __attribute__((ext_vector_type(2)));
typedef float    v8f  __attribute__((ext_vector_type(8)));
typedef float    v4f  __attribute__((ext_vector_type(4)));
typedef int      v4i  __attribute__((vector_size(16)));
typedef unsigned int u32x4 __attribute__((vector_size(16)));
typedef int      i32x8 __attribute__((vector_size(32)));
typedef int      i32x4 __attribute__((vector_size(16)));

#define KTILE 64
#define NTILE 64
#define MTOT  128
#define LDSW  72   /* f16 elems per LDS row: 64 data + 8 pad = 144 B (16B-aligned rows) */

#if __has_builtin(__builtin_amdgcn_global_load_async_to_lds_b128)
#define USE_ASYNC 1
#else
#define USE_ASYNC 0
#endif

#if __has_builtin(__builtin_amdgcn_tensor_load_to_lds)
#define USE_TDM 1
#else
#define USE_TDM 0
#endif

#if __has_builtin(__builtin_amdgcn_s_wait_tensorcnt)
#define WAIT_TENSOR() __builtin_amdgcn_s_wait_tensorcnt(0)
#else
#define WAIT_TENSOR() asm volatile("s_wait_tensorcnt 0x0" ::: "memory")
#endif

union U16 { v16h v; v8h h[2]; };

static __device__ __forceinline__ h2 cvt2(float a, float b) {
#if __has_builtin(__builtin_amdgcn_cvt_pkrtz)
    return __builtin_bit_cast(h2, __builtin_amdgcn_cvt_pkrtz(a, b));
#else
    h2 r; r.x = (_Float16)a; r.y = (_Float16)b; return r;
#endif
}

#if USE_TDM
// TDM: DMA the 128x64 f16 X tile (row-major [128,K]) into LDS with 16B row padding.
// D# per CDNA5 ISA 8.3/8.4: count=1, type=2(image); data_size=2B; pad 4 DWORDs per 32.
static __device__ __forceinline__ void tdm_load_x(const _Float16* xh, unsigned ldsOff,
                                                  int K, int kBase) {
    unsigned long long ga = (unsigned long long)(size_t)(xh + kBase);
    u32x4 g0 = { 1u,                                   // count=1
                 ldsOff,                               // lds_addr
                 (unsigned)ga,                         // global_addr[31:0]
                 (unsigned)((ga >> 32) & 0x01FFFFFFu) | (2u << 30) }; // addr[56:32]|type=2
    i32x8 g1 = { (int)((1u << 16) | (1u << 20) | (4u << 22) | (3u << 25)),
                 //     data_size=2B  pad_en      pad_int=32dw  pad_amt=4dw
                 (int)((unsigned)(K & 0xFFFF) << 16),              // tensor_dim0 lo16
                 (int)(((unsigned)K >> 16) | (128u << 16)),        // dim0 hi | dim1 lo
                 (int)(64u << 16),                                 // tile_dim0 = 64
                 128,                                              // tile_dim1 = 128
                 K,                                                // dim0_stride lo32
                 0, 0 };
    i32x4 z4 = { 0, 0, 0, 0 };
    i32x8 z8 = { 0, 0, 0, 0, 0, 0, 0, 0 };
    __builtin_amdgcn_tensor_load_to_lds(g0, g1, z4, z4, z8, 0);
}
#endif

// ---- X tile f16 fallback staging (no TDM): per-lane copy ----
static __device__ __forceinline__ void stage_x_f16(const _Float16* __restrict__ xh,
                                                   _Float16* xbuf, int K, int kBase, int t) {
    #pragma unroll
    for (int i = 0; i < 4; ++i) {
        int f = i * 256 + t;
        int m = f >> 3, kq = f & 7;
        const _Float16* src = xh + (size_t)m * K + kBase + kq * 8;
        _Float16*       dst = xbuf + m * LDSW + kq * 8;
#if USE_ASYNC
        __builtin_amdgcn_global_load_async_to_lds_b128((v4i*)src, (v4i*)dst, 0, 0);
#else
        *(v8h*)dst = *(const v8h*)src;
#endif
    }
}

// ---- X tile f32 fallback: batch loads, then convert+store ----
static __device__ __forceinline__ void stage_x_f32(const float* __restrict__ x,
                                                   _Float16* xbuf, int K, int kBase, int t) {
    v4f tmp[8];
    #pragma unroll
    for (int i = 0; i < 8; ++i) {
        int f = i * 256 + t; int m = f >> 4, kq = f & 15;
        tmp[i] = *(const v4f*)(x + (size_t)m * K + kBase + kq * 4);
    }
    #pragma unroll
    for (int i = 0; i < 8; ++i) {
        int f = i * 256 + t; int m = f >> 4, kq = f & 15;
        h2* dst = (h2*)&xbuf[m * LDSW + kq * 4];
        dst[0] = cvt2(tmp[i].x, tmp[i].y);
        dst[1] = cvt2(tmp[i].z, tmp[i].w);
    }
}

template <bool XF16>
__global__ __launch_bounds__(256) void gptq_wmma_kernel(
    const void*  __restrict__ xin,      // XF16 ? f16 [128,K] : f32 [128,K]
    const int*   __restrict__ qweight,  // [K/8, N]
    const int*   __restrict__ qzeros,   // [G, N/8]
    const float* __restrict__ scales,   // [G, N]
    const float* __restrict__ bias,     // [N]
    float*       __restrict__ out,      // [128, N]
    int K, int N)
{
    __shared__ _Float16 XT[2][MTOT * LDSW];     // activations f16
    __shared__ _Float16 WT[2][NTILE * LDSW];    // dequantized weights f16, [n][k]
    __shared__ int      QWB[2][8 * 64];         // packed int4 weight tile (async-staged)

    const int t    = threadIdx.x;
    const int wave = t >> 5;
    const int lane = t & 31;
    const int nBase = blockIdx.x * NTILE;

    v8f acc[4] = {};

    const int mRow = (wave << 4) + (lane & 15);
    const int aOff = (lane >> 4) << 3;   // A: 0 / 8
    const int bOff = (lane >> 4) << 3;   // B: 0 / 8

    // phase 1: kick off bulk data movement for a buffer
    auto issue_stage = [&](int buf, int kBase) {
        if (XF16) {
#if USE_TDM
            if (wave == 0)
                tdm_load_x((const _Float16*)xin, (unsigned)(size_t)&XT[buf][0], K, kBase);
#else
            stage_x_f16((const _Float16*)xin, XT[buf], K, kBase, t);
#endif
        } else {
            stage_x_f32((const float*)xin, XT[buf], K, kBase, t);
        }
        // packed qweight tile: 8 rows x 64 int32 = 2KB, async DMA to LDS (waves 0-3)
        if (t < 128) {
            int kp = t >> 4, c = (t & 15) * 4;
            const int* src = qweight + (size_t)((kBase >> 3) + kp) * N + nBase + c;
            int*       dst = &QWB[buf][kp * 64 + c];
#if USE_ASYNC
            __builtin_amdgcn_global_load_async_to_lds_b128((v4i*)src, (v4i*)dst, 0, 0);
#else
            *(v4i*)dst = *(const v4i*)src;
#endif
        }
    };

    // phase 2 (after s_wait_asynccnt): LDS->LDS dequant int4 -> f16
    auto dequant_stage = [&](int buf, int kBase) {
        const int g = kBase >> 7;   // GROUP = 128
        unsigned qw[2], qz[2]; float sc[2]; int kp[2], nn[2];
        #pragma unroll
        for (int e2 = 0; e2 < 2; ++e2) {
            int e = e2 * 256 + t;
            kp[e2] = e >> 6; nn[e2] = e & 63;
            int nG = nBase + nn[e2];
            qw[e2] = (unsigned)QWB[buf][kp[e2] * 64 + nn[e2]];
            sc[e2] = scales[(size_t)g * N + nG];
            qz[e2] = (unsigned)qzeros[(size_t)g * (N >> 3) + (nG >> 3)];
        }
        #pragma unroll
        for (int e2 = 0; e2 < 2; ++e2) {
            int nG = nBase + nn[e2];
            float zf = (float)((qz[e2] >> ((nG & 7) * 4)) & 0xF) + 1.0f;
            union { v8h v; h2 p[4]; } wv;
            #pragma unroll
            for (int i = 0; i < 4; ++i) {
                float a = sc[e2] * ((float)((qw[e2] >> (8 * i    )) & 0xF) - zf);
                float b = sc[e2] * ((float)((qw[e2] >> (8 * i + 4)) & 0xF) - zf);
                wv.p[i] = cvt2(a, b);
            }
            *(v8h*)&WT[buf][nn[e2] * LDSW + kp[e2] * 8] = wv.v;
        }
    };

    const int nIter = K / KTILE;

    // prologue: fully stage buffer 0
    issue_stage(0, 0);
#if USE_ASYNC
    asm volatile("s_wait_asynccnt 0x0" ::: "memory");
#endif
    dequant_stage(0, 0);
#if USE_TDM
    if (XF16) WAIT_TENSOR();
#endif
    __syncthreads();

    for (int it = 0; it < nIter; ++it) {
        const int cur = it & 1;
        const bool hasNext = (it + 1 < nIter);

        if (hasNext) issue_stage(cur ^ 1, (it + 1) * KTILE);
        if (it + 2 < nIter)
            __builtin_prefetch(qweight + (size_t)(((it + 2) * KTILE) >> 3) * N + nBase + (t & 63), 0, 1);

        // ---- matrix core: software-pipelined fragments ----
        const _Float16* xrow  = &XT[cur][mRow * LDSW];
        const _Float16* wbase = &WT[cur][(lane & 15) * LDSW];
        U16 afr[2]; U16 bfr[2][4];
        auto ldfrag = [&](int kk, int slot) {
            afr[slot].h[0] = *(const v8h*)(xrow + kk + aOff);
            afr[slot].h[1] = *(const v8h*)(xrow + kk + 16 + aOff);
            #pragma unroll
            for (int s = 0; s < 4; ++s) {
                const _Float16* wrow = wbase + s * 16 * LDSW;
                bfr[slot][s].h[0] = *(const v8h*)(wrow + kk + bOff);
                bfr[slot][s].h[1] = *(const v8h*)(wrow + kk + bOff + 16);
            }
        };
        ldfrag(0, 0);
        #pragma unroll
        for (int kk2 = 0; kk2 < KTILE / 32; ++kk2) {
            if (kk2 + 1 < KTILE / 32) ldfrag((kk2 + 1) * 32, (kk2 + 1) & 1);
            #pragma unroll
            for (int s = 0; s < 4; ++s) {
                acc[s] = __builtin_amdgcn_wmma_f32_16x16x32_f16(
                    false, afr[kk2 & 1].v, false, bfr[kk2 & 1][s].v,
                    (short)0, acc[s], false, false);
            }
        }

        if (hasNext) {
#if USE_ASYNC
            asm volatile("s_wait_asynccnt 0x0" ::: "memory");
#endif
            dequant_stage(cur ^ 1, (it + 1) * KTILE);
        }
#if USE_TDM
        if (XF16) WAIT_TENSOR();
#endif
        __syncthreads();
    }

    // ---- epilogue: C/D layout -> out (non-temporal), add bias ----
    const int mOut = (wave << 4) + ((lane >> 4) << 3);
    #pragma unroll
    for (int s = 0; s < 4; ++s) {
        int col = nBase + s * 16 + (lane & 15);
        float bv = bias[col];
        #pragma unroll
        for (int r = 0; r < 8; ++r) {
            __builtin_nontemporal_store(acc[s][r] + bv, &out[(size_t)(mOut + r) * N + col]);
        }
    }
}

// ---- pre-pass: x f32 -> f16 into workspace (8 elems/thread) ----
__global__ __launch_bounds__(256) void cvt_x_kernel(const float* __restrict__ x,
                                                    _Float16* __restrict__ xh, int n) {
    int i = (blockIdx.x * 256 + threadIdx.x) * 8;
    if (i + 8 <= n) {
        v4f a = *(const v4f*)(x + i);
        v4f b = *(const v4f*)(x + i + 4);
        union { v8h v; h2 p[4]; } r;
        r.p[0] = cvt2(a.x, a.y); r.p[1] = cvt2(a.z, a.w);
        r.p[2] = cvt2(b.x, b.y); r.p[3] = cvt2(b.z, b.w);
        *(v8h*)(xh + i) = r.v;
    }
}

extern "C" void kernel_launch(void* const* d_in, const int* in_sizes, int n_in,
                              void* d_out, int out_size, void* d_ws, size_t ws_size,
                              hipStream_t stream) {
    const float* x       = (const float*)d_in[0];
    const int*   qweight = (const int*)  d_in[1];
    const int*   qzeros  = (const int*)  d_in[2];
    const float* scales  = (const float*)d_in[3];
    const float* bias    = (const float*)d_in[4];
    float*       out     = (float*)d_out;

    const int N = in_sizes[4];                         // 14336
    const int K = (int)((long)in_sizes[1] * 8 / N);    // 4096
    const size_t xElems = (size_t)in_sizes[0];         // 128*K

    dim3 grid(N / NTILE), block(256);

    if (ws_size >= xElems * sizeof(_Float16)) {
        _Float16* xh = (_Float16*)d_ws;
        int nblk = (int)((xElems / 8 + 255) / 256);
        cvt_x_kernel<<<nblk, 256, 0, stream>>>(x, xh, (int)xElems);
        gptq_wmma_kernel<true><<<grid, block, 0, stream>>>(xh, qweight, qzeros, scales, bias, out, K, N);
    } else {
        gptq_wmma_kernel<false><<<grid, block, 0, stream>>>(x, qweight, qzeros, scales, bias, out, K, N);
    }
}
